// LlamaDecoderLayer_91336774517629
// MI455X (gfx1250) — matrix operand order repaired and checked by reading, not compiled, not run on hardware
//
#include <hip/hip_runtime.h>
#include <stdint.h>

#define H   4096
#define NH  32
#define NKV 8
#define HD  128
#define FFI 14336
#define SEQ 2048
#define QKV_O ((NH + 2*NKV)*HD)   /* 6144 */
#define ATT_SCALE 0.08838834764831845f  /* 128^-0.5 */
#define LN_THETA  13.122363377404328f   /* ln(500000) */

typedef __attribute__((ext_vector_type(16))) __bf16 v16bf;
typedef __attribute__((ext_vector_type(8)))  float  v8f;
typedef __attribute__((ext_vector_type(4)))  float  v4f;
typedef __attribute__((ext_vector_type(8)))  unsigned int v8u;
typedef __attribute__((ext_vector_type(4)))  unsigned int v4u;
typedef __attribute__((ext_vector_type(8)))  int v8i;
typedef __attribute__((ext_vector_type(4)))  int v4i;

// ---------- helpers ----------
__device__ __forceinline__ unsigned pack_bf16(float a, float b) {
  unsigned ua = __builtin_bit_cast(unsigned, a);
  unsigned ub = __builtin_bit_cast(unsigned, b);
  ua += 0x7FFFu + ((ua >> 16) & 1u);           // RNE
  ub += 0x7FFFu + ((ub >> 16) & 1u);
  return (ua >> 16) | (ub & 0xFFFF0000u);
}
__device__ __forceinline__ __bf16 f2bf(float a) {
  unsigned ua = __builtin_bit_cast(unsigned, a);
  ua += 0x7FFFu + ((ua >> 16) & 1u);
  unsigned short s = (unsigned short)(ua >> 16);
  return __builtin_bit_cast(__bf16, s);
}
__device__ __forceinline__ float bf2f(__bf16 x) {
  unsigned u = ((unsigned)__builtin_bit_cast(unsigned short, x)) << 16;
  return __builtin_bit_cast(float, u);
}
__device__ __forceinline__ v8f wmma_bf16(v8u a, v8u b, v8f c) {
  return __builtin_amdgcn_wmma_f32_16x16x32_bf16(
      false, __builtin_bit_cast(v16bf, a),
      false, __builtin_bit_cast(v16bf, b),
      (short)0, c, false, false);
}
__device__ __forceinline__ v8f vzero8() {
  v8f z;
#pragma unroll
  for (int i = 0; i < 8; ++i) z[i] = 0.f;
  return z;
}
// 8 contiguous dwords (32B) as two b128 loads
__device__ __forceinline__ v8u load8(const unsigned* p) {
  v4u a = *(const v4u*)p, b = *(const v4u*)(p + 4);
  v8u r;
  r[0]=a[0]; r[1]=a[1]; r[2]=a[2]; r[3]=a[3];
  r[4]=b[0]; r[5]=b[1]; r[6]=b[2]; r[7]=b[3];
  return r;
}
// two disjoint 16B groups as two b128 loads
__device__ __forceinline__ v8u load44(const unsigned* p0, const unsigned* p1) {
  v4u a = *(const v4u*)p0, b = *(const v4u*)p1;
  v8u r;
  r[0]=a[0]; r[1]=a[1]; r[2]=a[2]; r[3]=a[3];
  r[4]=b[0]; r[5]=b[1]; r[6]=b[2]; r[7]=b[3];
  return r;
}
__device__ __forceinline__ void store_val(float*  p, float v) { *p = v; }
__device__ __forceinline__ void store_val(__bf16* p, float v) { *p = f2bf(v); }

#if __has_builtin(__builtin_amdgcn_tensor_load_to_lds)
#define USE_TDM 1
// Build a 2D Tensor-DMA descriptor (D#) for a 16-bit tile and issue TENSOR_LOAD_TO_LDS.
// Layout per CDNA5 ISA 8.3/8.4: g0 = {flags(count=1), lds_addr, gaddr_lo, gaddr_hi|type=2},
// g1 packs data_size=2B, tensor_dim0/1, tile_dim0/1, tensor_dim0_stride.
// 6-arg toolchain form: (v4u, v8i, v4i, v4i, v8i, i32 cpol).
__device__ __forceinline__ void tdm_load_tile_b16(const void* g, unsigned lds_off,
                                                  unsigned tile_w, unsigned tile_h,
                                                  unsigned stride_elems,
                                                  unsigned dim0, unsigned dim1) {
  unsigned long long ga = (unsigned long long)g;
  v4u g0;
  g0[0] = 1u;                                                  // count=1, user mode
  g0[1] = lds_off;                                             // lds_addr (bytes)
  g0[2] = (unsigned)ga;                                        // global_addr[31:0]
  g0[3] = (unsigned)((ga >> 32) & 0x01FFFFFFu) | 0x80000000u;  // addr[56:32] | type=2
  v8i g1;
  g1[0] = (int)(1u << 16);                                     // data_size code 1 = 2 bytes
  g1[1] = (int)((dim0 & 0xFFFFu) << 16);                       // tensor_dim0[15:0]
  g1[2] = (int)((dim0 >> 16) | ((dim1 & 0xFFFFu) << 16));      // dim0[31:16] | dim1[15:0]
  g1[3] = (int)((dim1 >> 16) | (tile_w << 16));                // dim1[31:16] | tile_dim0
  g1[4] = (int)(tile_h & 0xFFFFu);                             // tile_dim1 (tile_dim2=0)
  g1[5] = (int)stride_elems;                                   // tensor_dim0_stride[31:0]
  g1[6] = 0;
  g1[7] = 0;
  v4i z4; z4[0] = z4[1] = z4[2] = z4[3] = 0;
  v8i z8;
#pragma unroll
  for (int i = 0; i < 8; ++i) z8[i] = 0;
  __builtin_amdgcn_tensor_load_to_lds(g0, g1, z4, z4, z8, 0);
}
__device__ __forceinline__ void tdm_wait0() {
#if __has_builtin(__builtin_amdgcn_s_wait_tensorcnt)
  __builtin_amdgcn_s_wait_tensorcnt(0);
#else
  asm volatile("s_wait_tensorcnt 0x0" ::: "memory");
#endif
}
#endif

// ---------- 1. fused (add +) RMSNorm -> bf16 activations ----------
template<bool ADD>
__global__ __launch_bounds__(256)
void rmsnorm_kernel(const float* __restrict__ x, const float* __restrict__ rin,
                    const float* __restrict__ w, float* __restrict__ rout,
                    __bf16* __restrict__ hout) {
  const int s = blockIdx.x, t = threadIdx.x;
  const size_t base = (size_t)s * H;
  float vals[16]; float ss = 0.f;
#pragma unroll
  for (int i = 0; i < 16; ++i) {
    int col = t + 256*i;
    float v = x[base + col];
    if (ADD) v += rin[base + col];
    vals[i] = v; ss += v*v;
  }
  __shared__ float red[256];
  red[t] = ss; __syncthreads();
  for (int off = 128; off > 0; off >>= 1) { if (t < off) red[t] += red[t+off]; __syncthreads(); }
  float inv = rsqrtf(red[0] * (1.0f/H) + 1e-5f);
#pragma unroll
  for (int i = 0; i < 16; ++i) {
    int col = t + 256*i;
    float v = vals[i];
    if (ADD) rout[base + col] = v;
    hout[base + col] = f2bf(v * inv * w[col]);
  }
}

// ---------- 2. generic WMMA GEMM: C[M,N] = A[M,K](bf16) * W[N,K](f32,cast)^T ----------
// block tile 128(M) x 64(N), BK=32, 256 threads = 8 waves -> each wave 32x32 (2x2 wmma)
// A tile staged via Tensor Data Mover (async DMA) when available.
template<typename OUT_T, bool ADD_RES>
__global__ __launch_bounds__(256)
void gemm_bf16_wmma(const __bf16* __restrict__ A, const float* __restrict__ W,
                    OUT_T* __restrict__ C, const float* __restrict__ res,
                    int M, int N, int K) {
  __shared__ __align__(16) unsigned As[128*16];  // [m][k_pair]  (8 KB)
  __shared__ __align__(16) unsigned Ws[64*16];   // [n][k_pair]  (4 KB)
  const int t = threadIdx.x, lane = t & 31, wid = t >> 5;
  const int hl = lane >> 4, l16 = lane & 15;
  const int wm = wid & 3, wn = wid >> 2;
  const int m0 = blockIdx.y * 128, n0 = blockIdx.x * 64;
  const unsigned* Au = (const unsigned*)A;
  (void)Au;

  v8f acc[2][2];
#pragma unroll
  for (int i = 0; i < 2; ++i)
#pragma unroll
    for (int j = 0; j < 2; ++j) acc[i][j] = vzero8();

  for (int k0 = 0; k0 < K; k0 += 32) {
#ifdef USE_TDM
    if (wid == 0) {  // async DMA the 128x32 bf16 A tile into LDS (TENSORcnt tracked)
      tdm_load_tile_b16((const char*)A + ((size_t)m0*K + k0)*2,
                        (unsigned)(unsigned long long)(void*)As,
                        /*tile_w=*/32u, /*tile_h=*/128u,
                        /*stride=*/(unsigned)K,
                        /*dim0=*/(unsigned)K, /*dim1=*/(unsigned)M);
    }
#else
    { // stage A tile (bf16 pairs, already packed): two b128 each
      int m = t >> 1, uc = (t & 1) * 8;
      const unsigned* src = Au + (size_t)(m0 + m) * (K >> 1) + (k0 >> 1) + uc;
      v4u a = *(const v4u*)src, b = *(const v4u*)(src + 4);
      *(v4u*)&As[m*16 + uc]     = a;
      *(v4u*)&As[m*16 + uc + 4] = b;
    }
#endif
    { // stage W tile, fp32 -> bf16 conversion on the fly, 2x b128 loads
      int n = t >> 2, fc = (t & 3) * 8;
      const float* srcf = W + (size_t)(n0 + n) * K + k0 + fc;
      if (k0 + 32 < K) __builtin_prefetch(srcf + 32, 0, 0);   // global_prefetch_b8
      v4f x = *(const v4f*)srcf, y = *(const v4f*)(srcf + 4);
      int base = n*16 + (t & 3)*4;
      Ws[base+0] = pack_bf16(x[0], x[1]);
      Ws[base+1] = pack_bf16(x[2], x[3]);
      Ws[base+2] = pack_bf16(y[0], y[1]);
      Ws[base+3] = pack_bf16(y[2], y[3]);
    }
#ifdef USE_TDM
    if (wid == 0) tdm_wait0();   // s_wait_tensorcnt 0 before sharing the tile
#endif
    __syncthreads();
    v8u aU[2], bU[2];
#pragma unroll
    for (int i = 0; i < 2; ++i) {          // A frag: two ds_load_b128
      int m = wm*32 + i*16 + l16;
      aU[i] = load44(&As[m*16 + hl*4], &As[m*16 + hl*4 + 8]);
    }
#pragma unroll
    for (int j = 0; j < 2; ++j) {          // B frag: two 4-dword groups (two ds_load_b128)
      int n = wn*32 + j*16 + l16;
      bU[j] = load44(&Ws[n*16 + hl*4], &Ws[n*16 + hl*4 + 8]);
    }
#pragma unroll
    for (int i = 0; i < 2; ++i)
#pragma unroll
      for (int j = 0; j < 2; ++j)
        acc[i][j] = wmma_bf16(aU[i], bU[j], acc[i][j]);
    __syncthreads();
  }
  // epilogue: C layout -> lane=N col, vgpr r = row (r + 8*half)
#pragma unroll
  for (int i = 0; i < 2; ++i)
#pragma unroll
    for (int j = 0; j < 2; ++j)
#pragma unroll
      for (int r = 0; r < 8; ++r) {
        int row = m0 + wm*32 + i*16 + r + 8*hl;
        int col = n0 + wn*32 + j*16 + l16;
        float v = acc[i][j][r];
        if (ADD_RES) v += res[(size_t)row * N + col];
        store_val(&C[(size_t)row * N + col], v);
      }
}

// ---------- 3a. RoPE + split QKV into bf16 Q/K ----------
__global__ __launch_bounds__(256)
void rope_split_kernel(const float* __restrict__ qkv, const int* __restrict__ pos,
                       __bf16* __restrict__ q, __bf16* __restrict__ k) {
  const int s = blockIdx.x, t = threadIdx.x;
  const float p = (float)pos[s];
  const float* row = qkv + (size_t)s * QKV_O;
  for (int idx = t; idx < NH*64; idx += 256) {          // Q heads
    int h = idx >> 6, j = idx & 63;
    float inv = __expf(-LN_THETA * (float)j * (1.0f/64.f));
    float sn, cs; __sincosf(p * inv, &sn, &cs);
    float x1 = row[h*HD + j], x2 = row[h*HD + 64 + j];
    q[((size_t)s*NH + h)*HD + j]      = f2bf(x1*cs - x2*sn);
    q[((size_t)s*NH + h)*HD + 64 + j] = f2bf(x2*cs + x1*sn);
  }
  const float* krow = row + NH*HD;
  for (int idx = t; idx < NKV*64; idx += 256) {         // K heads
    int h = idx >> 6, j = idx & 63;
    float inv = __expf(-LN_THETA * (float)j * (1.0f/64.f));
    float sn, cs; __sincosf(p * inv, &sn, &cs);
    float x1 = krow[h*HD + j], x2 = krow[h*HD + 64 + j];
    k[((size_t)s*NKV + h)*HD + j]      = f2bf(x1*cs - x2*sn);
    k[((size_t)s*NKV + h)*HD + 64 + j] = f2bf(x2*cs + x1*sn);
  }
}

// ---------- 3b. tiled transpose of V: qkv[:, (NH+NKV)*HD + d] -> vt[d][s] (bf16) ----------
// vt layout: [NKV*HD][SEQ]; makes P.V B-fragments contiguous along the key axis.
__global__ __launch_bounds__(256)
void v_transpose_kernel(const float* __restrict__ qkv, __bf16* __restrict__ vt) {
  __shared__ float tile[32][33];
  const int s0 = blockIdx.x * 32, d0 = blockIdx.y * 32;
  const int tx = threadIdx.x & 31, ty = threadIdx.x >> 5;   // 32 x 8
#pragma unroll
  for (int r = 0; r < 32; r += 8) {
    int s = s0 + ty + r;
    tile[ty + r][tx] = qkv[(size_t)s*QKV_O + (NH + NKV)*HD + d0 + tx];
  }
  __syncthreads();
#pragma unroll
  for (int r = 0; r < 32; r += 8) {
    int d = d0 + ty + r;
    vt[(size_t)d*SEQ + s0 + tx] = f2bf(tile[tx][ty + r]);
  }
}

// ---------- 4. causal flash attention (GQA), all matmuls via WMMA ----------
// grid (SEQ/128, NH); 8 waves/block; wave w owns q-tile (blockIdx.x*8+w): 16 queries.
__global__ __launch_bounds__(256)
void flash_attn_kernel(const __bf16* __restrict__ Q, const __bf16* __restrict__ Kc,
                       const __bf16* __restrict__ Vt, __bf16* __restrict__ O) {
  __shared__ __align__(16) __bf16 Plds[8][16][32];   // per-wave P staging (8 KB)
  const int t = threadIdx.x, lane = t & 31, wid = t >> 5;
  const int hl = lane >> 4, l16 = lane & 15;
  const int h = blockIdx.y, kvh = h >> 2;
  const int qt = blockIdx.x * 8 + wid;
  const unsigned* Qu = (const unsigned*)Q;
  const unsigned* Ku = (const unsigned*)Kc;
  const unsigned* Vu = (const unsigned*)Vt;

  // Q fragments: 16 rows x 128 (4 chunks of K=32), each = two b128 loads
  v8u qf[4];
  const int qtok = qt*16 + l16;
  const unsigned* qrowp = Qu + ((size_t)qtok*NH + h)*(HD/2);
#pragma unroll
  for (int c = 0; c < 4; ++c)
    qf[c] = load44(qrowp + c*16 + hl*4, qrowp + c*16 + hl*4 + 8);

  float m_run[8], l_run[8];
  v8f acc[8];
#pragma unroll
  for (int r = 0; r < 8; ++r) { m_run[r] = -1e30f; l_run[r] = 0.f; }
#pragma unroll
  for (int nt = 0; nt < 8; ++nt) acc[nt] = vzero8();

  const int nkb = (qt >> 1) + 1;          // 32-key chunks needed (causal)
  for (int kb = 0; kb < nkb; ++kb) {
    const int key0 = kb * 32;
    // ---- S = Q K^T over this 32-key chunk (two 16-col tiles) ----
    const unsigned* k0p = Ku + ((size_t)(key0      + l16)*NKV + kvh)*(HD/2);
    const unsigned* k1p = Ku + ((size_t)(key0 + 16 + l16)*NKV + kvh)*(HD/2);
    v8f s0 = vzero8(), s1 = vzero8();
#pragma unroll
    for (int c = 0; c < 4; ++c) {
      v8u b0 = load44(k0p + c*16 + hl*4, k0p + c*16 + hl*4 + 8);   // two 4-dword groups along head dim
      v8u b1 = load44(k1p + c*16 + hl*4, k1p + c*16 + hl*4 + 8);
      s0 = wmma_bf16(qf[c], b0, s0);
      s1 = wmma_bf16(qf[c], b1, s1);
    }
    // ---- mask + online softmax (fully unrolled: all vector subscripts constant) ----
#pragma unroll
    for (int r = 0; r < 8; ++r) {
      int qrow = qt*16 + r + 8*hl;
      float v0 = s0[r] * ATT_SCALE, v1 = s1[r] * ATT_SCALE;
      if (key0 + l16      > qrow) v0 = -1e30f;
      if (key0 + 16 + l16 > qrow) v1 = -1e30f;
      float mx = fmaxf(v0, v1);
#pragma unroll
      for (int off = 8; off > 0; off >>= 1) mx = fmaxf(mx, __shfl_xor(mx, off));
      float mnew = fmaxf(m_run[r], mx);
      float corr = __expf(m_run[r] - mnew);
      float p0 = __expf(v0 - mnew), p1 = __expf(v1 - mnew);
      float rs = p0 + p1;
#pragma unroll
      for (int off = 8; off > 0; off >>= 1) rs += __shfl_xor(rs, off);
      l_run[r] = l_run[r] * corr + rs;
      m_run[r] = mnew;
#pragma unroll
      for (int nt = 0; nt < 8; ++nt) acc[nt][r] *= corr;
      int m = r + 8*hl;
      Plds[wid][m][l16]      = f2bf(p0);
      Plds[wid][m][16 + l16] = f2bf(p1);
    }
    asm volatile("s_wait_dscnt 0x0" ::: "memory");   // wave-internal LDS transpose visibility
    // ---- rebuild P as A-fragment (16x32): two ds_load_b128 ----
    const unsigned* pp = (const unsigned*)&Plds[wid][l16][0];
    v8u pf = load44(pp + hl*4, pp + hl*4 + 8);
    // ---- O += P V (B-fragment contiguous thanks to transposed V) ----
#pragma unroll
    for (int nt = 0; nt < 8; ++nt) {
      int d = nt*16 + l16;
      const unsigned* vp = Vu + ((size_t)(kvh*HD + d))*(SEQ/2) + key0/2 + hl*4;
      v8u bv = load44(vp, vp + 8);
      acc[nt] = wmma_bf16(pf, bv, acc[nt]);
    }
  }
  // ---- normalize + write ctx (bf16, [S, NH, HD]) ----
#pragma unroll
  for (int r = 0; r < 8; ++r) {
    float inv = 1.f / l_run[r];
    int stok = qt*16 + r + 8*hl;
#pragma unroll
    for (int nt = 0; nt < 8; ++nt)
      O[((size_t)stok*NH + h)*HD + nt*16 + l16] = f2bf(acc[nt][r] * inv);
  }
}

// ---------- 5. SiLU(gate) * up ----------
__global__ __launch_bounds__(256)
void silu_mul_kernel(const __bf16* __restrict__ gu, __bf16* __restrict__ act) {
  const int s = blockIdx.y;
  const int i = blockIdx.x * 256 + threadIdx.x;
  float g = bf2f(gu[(size_t)s*(2*FFI) + i]);
  float u = bf2f(gu[(size_t)s*(2*FFI) + FFI + i]);
  act[(size_t)s*FFI + i] = f2bf(g / (1.f + __expf(-g)) * u);
}

// ---------- launcher ----------
extern "C" void kernel_launch(void* const* d_in, const int* in_sizes, int n_in,
                              void* d_out, int out_size, void* d_ws, size_t ws_size,
                              hipStream_t stream) {
  (void)in_sizes; (void)n_in; (void)out_size; (void)ws_size;
  const int*   pos   = (const int*)  d_in[0];
  const float* hid   = (const float*)d_in[1];
  const float* resin = (const float*)d_in[2];
  const float* wqkv  = (const float*)d_in[3];
  const float* wo    = (const float*)d_in[4];
  const float* wgu   = (const float*)d_in[5];
  const float* wdown = (const float*)d_in[6];
  const float* wln1  = (const float*)d_in[7];
  const float* wln2  = (const float*)d_in[8];

  float* out_f = (float*)d_out;                 // output 0: layer out  [S,H]
  float* res_f = out_f + (size_t)SEQ * H;       // output 1: residual   [S,H]

  char* w = (char*)d_ws;
  const size_t MB = 1ull << 20;
  __bf16* h1  = (__bf16*)(w + 0*MB);     // 16 MB  normed activations 1
  float*  qkv = (float*) (w + 16*MB);    // 48 MB  qkv projection (pre-rope)
  __bf16* qb  = (__bf16*)(w + 64*MB);    // 16 MB
  __bf16* kb  = (__bf16*)(w + 80*MB);    //  4 MB
  __bf16* vt  = (__bf16*)(w + 84*MB);    //  4 MB  V transposed [NKV*HD][SEQ]
  __bf16* ctx = (__bf16*)(w + 88*MB);    // 16 MB
  __bf16* h2  = (__bf16*)(w + 104*MB);   // 16 MB  normed activations 2
  __bf16* gu  = (__bf16*)(w + 120*MB);   // 112 MB gate|up
  __bf16* act = (__bf16*)(w + 232*MB);   // 56 MB  silu(gate)*up
  float*  r1  = (float*) (w + 288*MB);   // 32 MB  residual after first add

  rmsnorm_kernel<true ><<<SEQ, 256, 0, stream>>>(hid, resin, wln1, r1, h1);
  gemm_bf16_wmma<float, false><<<dim3(QKV_O/64, SEQ/128), 256, 0, stream>>>(
      h1, wqkv, qkv, nullptr, SEQ, QKV_O, H);
  rope_split_kernel<<<SEQ, 256, 0, stream>>>(qkv, pos, qb, kb);
  v_transpose_kernel<<<dim3(SEQ/32, (NKV*HD)/32), 256, 0, stream>>>(qkv, vt);
  flash_attn_kernel<<<dim3(SEQ/128, NH), 256, 0, stream>>>(qb, kb, vt, ctx);
  gemm_bf16_wmma<float, true ><<<dim3(H/64, SEQ/128), 256, 0, stream>>>(
      ctx, wo, res_f, r1, SEQ, H, H);                       // attn_out + residual
  rmsnorm_kernel<false><<<SEQ, 256, 0, stream>>>(res_f, nullptr, wln2, nullptr, h2);
  gemm_bf16_wmma<__bf16, false><<<dim3((2*FFI)/64, SEQ/128), 256, 0, stream>>>(
      h2, wgu, gu, nullptr, SEQ, 2*FFI, H);
  silu_mul_kernel<<<dim3(FFI/256, SEQ), 256, 0, stream>>>(gu, act);
  gemm_bf16_wmma<float, false><<<dim3(H/64, SEQ/128), 256, 0, stream>>>(
      act, wdown, out_f, nullptr, SEQ, H, FFI);
}
